// MPS_RNN_2D_81466939670972
// MI455X (gfx1250) — matrix operand order repaired and checked by reading, not compiled, not run on hardware
//
#include <hip/hip_runtime.h>
#include <hip/hip_bf16.h>
#include <math.h>

typedef _Float16 v16h __attribute__((ext_vector_type(16)));
typedef __fp16   fp16x2 __attribute__((ext_vector_type(2)));
typedef float    v8f  __attribute__((ext_vector_type(8)));

#define EPSF 1e-10f
#define PI_F 3.14159265358979323846f
#define HALF_LN2_F 0.34657359027997264f   // 0.5 * ln(2)
#define WF_BYTES (64 * 8 * 32 * 32)       // 512 KB prepped weight A-fragments
#define E2_BYTES (8 * 8 * 32 * 4)         // 8 KB eta^2 (lattice order)

// lane^16 exchange as a single VALU op: v_permlanex16_b32 with identity
// nibble selects (each lane reads its own position in the other 16-lane row).
__device__ __forceinline__ int xor16_i(int x) {
    return __builtin_amdgcn_permlanex16(x, x, 0x76543210, (int)0xfedcba98,
                                        false, false);
}
__device__ __forceinline__ float xor16_f(float x) {
    union { float f; int i; } a, b;
    a.f = x; b.i = xor16_i(a.i); return b.f;
}

// ---------------------------------------------------------------------------
// Prep 1: swizzle Wt = transpose(W[64][64]) per site into WMMA **A**-fragment
// layout, f32 -> f16.
//   W[k][n]: n = s*32+a ; k<32 -> M_h[i][jl][s][a][k], k>=32 -> M_v[...][k-32]
//   A layout (ISA 7.12.2): lane: row m = l&15; halves 0..7 -> Klocal = g*8+j,
//   halves 8..15 -> Klocal = 16+g*8+(j-8), g = l>>4.
// ---------------------------------------------------------------------------
__global__ void mps_prep_weights(const float* __restrict__ Mh,
                                 const float* __restrict__ Mv,
                                 _Float16* __restrict__ wf) {
    int site = blockIdx.x;               // scan order: site = i*8 + jscan
    int i  = site >> 3;
    int js = site & 7;
    int jl = (i & 1) ? (7 - js) : js;    // lattice column (snake)
    int f    = threadIdx.x >> 5;         // frag 0..7
    int lane = threadIdx.x & 31;
    int t = f >> 2, u = f & 3;
    int n = u * 16 + (lane & 15);
    int s = n >> 5, a = n & 31;
    int g8 = (lane >> 4) * 8;

    const float* MhB = Mh + ((((size_t)i * 8 + jl) * 2 + s) * 32 + a) * 32;
    const float* MvB = Mv + ((((size_t)i * 8 + jl) * 2 + s) * 32 + a) * 32;
    _Float16* dst = wf + ((size_t)(site * 8 + f) * 32 + lane) * 16;
    #pragma unroll
    for (int j = 0; j < 16; ++j) {
        int kl = (j < 8) ? (g8 + j) : (16 + g8 + (j - 8));
        int k  = t * 32 + kl;
        float val = (k < 32) ? MhB[k] : MvB[k - 32];
        dst[j] = (_Float16)val;
    }
}

// Prep 2: eta^2, same [L,M,D] lattice-order layout as eta.
__global__ void mps_prep_eta2(const float* __restrict__ eta,
                              float* __restrict__ e2) {
    int idx = blockIdx.x * 256 + threadIdx.x;   // 0..2047
    float t = eta[idx];
    e2[idx] = t * t;
}

// Prep 3: transpose x [B,64] int32 -> site-major bytes [64][B].
__global__ void mps_prep_x(const int* __restrict__ x,
                           unsigned char* __restrict__ xT, int B) {
    int idx  = blockIdx.x * 256 + threadIdx.x;   // idx = b*64 + site
    int b    = idx >> 6;
    int site = idx & 63;
    xT[(size_t)site * B + b] = (unsigned char)x[idx];
}

// ---------------------------------------------------------------------------
// Main: one wave32 owns 16 batch rows (batch = B-operand columns = lanes).
// Per site: 8 x v_wmma_f32_16x16x32_f16 computing hc^T = W^T @ [h_left;h_below]
// with the bias pre-loaded as the C operand; each lane holds its own batch
// row's outputs -> in-lane reductions + one permlanex16 combine per reduced
// quantity; in-lane fragment rebuild.
// ---------------------------------------------------------------------------
__global__ __launch_bounds__(256) void mps_rnn2d_main(
    const float* __restrict__ v,    // [L,M,S,D]
    const float* __restrict__ w,    // [L,M,D]
    const float* __restrict__ c,    // [L,M]
    const float* __restrict__ e2v,  // [L,M,D] eta^2
    const _Float16* __restrict__ wf,      // prepped weight A-fragments
    const unsigned char* __restrict__ xT, // [64][B] occupations
    float* __restrict__ out, int B) {     // [B,2]
    __shared__ v16h fragrow[8][8][32];    // 64 KB: per-wave vertical carry (B-frags)

    const int wave = threadIdx.x >> 5;
    const int lane = threadIdx.x & 31;
    const int g    = lane >> 4;     // half: owns hidden dims Ga=[g*8,g*8+8), Gb=16+Ga
    const int col  = lane & 15;     // batch column within tile
    const int b0   = (blockIdx.x * 8 + wave) * 16;

    v16h zf;
    #pragma unroll
    for (int q = 0; q < 16; ++q) zf[q] = (_Float16)0.0f;
    #pragma unroll
    for (int jl = 0; jl < 8; ++jl) fragrow[wave][jl][lane] = zf;   // bottom boundary

    float dla2 = 0.0f;   // accumulated in log2; scaled by 0.5*ln2 at the end
    float dphc = 0.0f;   // count of negative amplitudes; scaled by pi at the end

    // running pointers (advance by constant stride per site)
    const v16h* wfs = (const v16h*)wf + lane;           // + site*256
    const unsigned char* xp = xT + b0 + col;            // + site*B

    for (int i = 0; i < 8; ++i) {
        v16h F;                                   // left boundary = ones
        #pragma unroll
        for (int q = 0; q < 16; ++q) F[q] = (_Float16)1.0f;

        // snake: params advance +1 col on even rows, -1 on odd rows
        const int jl0   = (i & 1) ? 7 : 0;
        const int pstep = (i & 1) ? -1 : 1;
        int jl = jl0;
        const float* ep = e2v + (i * 8 + jl0) * 32 + g * 8;
        const float* wp = w   + (i * 8 + jl0) * 32 + g * 8;
        const float* vp = v   + (size_t)(i * 8 + jl0) * 64 + g * 8;
        const float* cp = c   + (i * 8 + jl0);

        for (int js = 0; js < 8; ++js) {
            v16h A1 = fragrow[wave][jl][lane];    // h_below B-fragment

            // site parameters for this lane's hidden-dim groups
            const float cc = *cp;
            float e2a[8], e2b[8], wga[8], wgb[8];
            #pragma unroll
            for (int r = 0; r < 8; ++r) {
                e2a[r] = ep[r];   e2b[r] = ep[16 + r];
                wga[r] = wp[r];   wgb[r] = wp[16 + r];
            }

            // hc^T[64 x 16] = W^T @ [h_left ; h_below] + bias (as C operand)
            v8f acc[4];
            #pragma unroll
            for (int u = 0; u < 4; ++u) {
                #pragma unroll
                for (int r = 0; r < 8; ++r) acc[u][r] = vp[u * 16 + r];
            }
            #pragma unroll
            for (int u = 0; u < 4; ++u) {
                v16h W0 = wfs[(u)     * 32];      // K-chunk 0 (h_left)
                v16h W1 = wfs[(4 + u) * 32];      // K-chunk 1 (h_below)
                acc[u] = __builtin_amdgcn_wmma_f32_16x16x32_f16(
                    false, W0, false, F,  (short)0, acc[u], false, false);
                acc[u] = __builtin_amdgcn_wmma_f32_16x16x32_f16(
                    false, W1, false, A1, (short)0, acc[u], false, false);
            }

            // eta-weighted squared norms (in-lane over this half's 16 dims)
            float p0 = 0.0f, p1 = 0.0f;
            #pragma unroll
            for (int r = 0; r < 8; ++r) {
                p0 += e2a[r] * acc[0][r] * acc[0][r] + e2b[r] * acc[1][r] * acc[1][r];
                p1 += e2a[r] * acc[2][r] * acc[2][r] + e2b[r] * acc[3][r] * acc[3][r];
            }

            // occupation for this lane's batch row
            const bool sel = *xp != 0;
            float hA[8], hB[8];
            #pragma unroll
            for (int r = 0; r < 8; ++r) {
                hA[r] = sel ? acc[2][r] : acc[0][r];   // dims a in Ga
                hB[r] = sel ? acc[3][r] : acc[1][r];   // dims a in Gb
            }
            float psel = sel ? p1 : p0;
            float ptot = p0 + p1;
            float n2 = 0.0f, raw = 0.0f;
            #pragma unroll
            for (int r = 0; r < 8; ++r) {
                n2  += hA[r] * hA[r] + hB[r] * hB[r];
                raw += hA[r] * wga[r] + hB[r] * wgb[r];
            }
            // combine the two lane-halves (each holds 16 of the 32 dims)
            psel += xor16_f(psel);
            ptot += xor16_f(ptot);
            n2   += xor16_f(n2);
            raw  += xor16_f(raw);

            float inv = __builtin_amdgcn_rcpf(__builtin_amdgcn_sqrtf(n2) + EPSF);
            dla2 += __builtin_amdgcn_logf(psel + EPSF)
                  - __builtin_amdgcn_logf(ptot + EPSF);   // log2; scaled later
            float aval = raw * inv + cc;
            dphc += (aval < 0.0f) ? 1.0f : 0.0f;

            // rebuild B-fragment of hn: lane needs Ga dims in halves 0..7, Gb dims in halves 8..15.
            // Both are held in-lane; the lane^16 copies xa/xb are not used.
            union HP { fp16x2 h; unsigned u; };
            unsigned pa[4], pbu[4], xa[4], xb[4];
            #pragma unroll
            for (int k = 0; k < 4; ++k) {
                HP t0, t1;
                t0.h = __builtin_amdgcn_cvt_pkrtz(hA[2 * k] * inv, hA[2 * k + 1] * inv);
                t1.h = __builtin_amdgcn_cvt_pkrtz(hB[2 * k] * inv, hB[2 * k + 1] * inv);
                pa[k]  = t0.u;
                pbu[k] = t1.u;
                xa[k] = (unsigned)xor16_i((int)pa[k]);
                xb[k] = (unsigned)xor16_i((int)pbu[k]);
            }
            union FV { unsigned u[8]; v16h vv; } nf;
            #pragma unroll
            for (int k = 0; k < 4; ++k) {
                nf.u[k]     = pa[k];    // halves 0..7  (a = g*8+q)
                nf.u[4 + k] = pbu[k];   // halves 8..15 (a = 16+g*8+(q-8))
            }
            F = nf.vv;                        // h_left for next site
            fragrow[wave][jl][lane] = F;      // h_below for next row

            // advance running pointers
            wfs += 256;                       // next site's weight fragments
            xp  += B;                         // next site's occupations
            jl  += pstep;
            ep  += pstep * 32;
            wp  += pstep * 32;
            vp  += pstep * 64;
            cp  += pstep;
        }
    }

    if (g == 0) {
        float2 res;
        res.x = HALF_LN2_F * dla2;
        res.y = PI_F * dphc;
        *(float2*)(out + (size_t)(b0 + col) * 2) = res;
    }
}

extern "C" void kernel_launch(void* const* d_in, const int* in_sizes, int n_in,
                              void* d_out, int out_size, void* d_ws, size_t ws_size,
                              hipStream_t stream) {
    const int*   x   = (const int*)d_in[0];
    const float* Mh  = (const float*)d_in[1];
    const float* Mv  = (const float*)d_in[2];
    const float* v   = (const float*)d_in[3];
    const float* w   = (const float*)d_in[4];
    const float* c   = (const float*)d_in[5];
    const float* eta = (const float*)d_in[6];
    float* out = (float*)d_out;

    _Float16*      wf  = (_Float16*)d_ws;                             // 512 KB
    float*         e2  = (float*)((char*)d_ws + WF_BYTES);            // 8 KB
    unsigned char* xT  = (unsigned char*)d_ws + WF_BYTES + E2_BYTES;  // 2 MB

    const int B = in_sizes[0] / 64;   // 32768

    mps_prep_weights<<<64, 256, 0, stream>>>(Mh, Mv, wf);
    mps_prep_eta2<<<8, 256, 0, stream>>>(eta, e2);
    mps_prep_x<<<(B * 64) / 256, 256, 0, stream>>>(x, xT, B);
    // one wave per 16 batch rows; 8 waves (256 threads) per block
    mps_rnn2d_main<<<B / 128, 256, 0, stream>>>(v, w, c, e2, wf, xT, out, B);
}
